// GBCNN_15960098472830
// MI455X (gfx1250) — compile-verified, matrix operand order checked by reading, not run on hardware
//
#include <hip/hip_runtime.h>
#include <hip/hip_bf16.h>
#include <stdint.h>

#define NN 100000      // nodes
#define NE 1600000     // edges
#define BN_EPS 1e-5f

typedef __attribute__((ext_vector_type(16))) __bf16 v16bf;
typedef __attribute__((ext_vector_type(8)))  float  v8f;

static inline long long llmin_host(long long a, long long b) { return a < b ? a : b; }

// ---------------------------------------------------------------- fill zero
__global__ void rgcn_fill0(float* __restrict__ p, long long n) {
  long long i = (long long)blockIdx.x * blockDim.x + threadIdx.x;
  long long stride = (long long)gridDim.x * blockDim.x;
  for (; i < n; i += stride) p[i] = 0.0f;
}

// ---------------------------------------------------------------- edge scatter-add
// agg[dst[e]][c] += x[src[e]][c]   (fp32 global atomics; working set is L2-resident)
__global__ void rgcn_scatter_add(const float* __restrict__ x,
                                 const long long* __restrict__ src,
                                 const long long* __restrict__ dst,
                                 float* __restrict__ agg,
                                 long long total, int C) {
  long long i = (long long)blockIdx.x * blockDim.x + threadIdx.x;
  long long stride = (long long)gridDim.x * blockDim.x;
  for (; i < total; i += stride) {
    long long e = i / C;
    int c = (int)(i - e * C);
    int s = (int)src[e];
    int d = (int)dst[e];
    float v = x[(long long)s * C + c];
    __hip_atomic_fetch_add(&agg[(long long)d * C + c], v,
                           __ATOMIC_RELAXED, __HIP_MEMORY_SCOPE_AGENT);
  }
}

// ---------------------------------------------------------------- f32 [N,K] -> bf16 [N,Kpad] (zero pad)
__global__ void rgcn_cvt_pad_bf16(const float* __restrict__ X, __bf16* __restrict__ Y,
                                  long long Nrows, int K, int Kpad) {
  long long totalPairs = Nrows * (Kpad >> 1);
  long long idx = (long long)blockIdx.x * blockDim.x + threadIdx.x;
  long long stride = (long long)gridDim.x * blockDim.x;
  unsigned int* __restrict__ Yp = (unsigned int*)Y;
  for (; idx < totalPairs; idx += stride) {
    long long i2 = idx << 1;
    long long n = i2 / Kpad;
    int k = (int)(i2 - n * Kpad);
    const float* xr = X + n * K;
    float v0 = (k < K) ? xr[k] : 0.0f;
    float v1 = (k + 1 < K) ? xr[k + 1] : 0.0f;
    union { unsigned int u; __bf16 h[2]; } pk;
    pk.h[0] = (__bf16)v0;
    pk.h[1] = (__bf16)v1;
    Yp[idx] = pk.u;
  }
}

// ---------------------------------------------------------------- pack weights into B-fragment order
// W [K,C] f32 -> P: for each (kblk, ntile): 32 lanes x 16 bf16, lane-contiguous.
// element e of lane: kk = kblk*32 + khalf + (e&7) + ((e>>3)<<4), col = ntile*16 + (lane&15)
__global__ void rgcn_pack_w(const float* __restrict__ W, __bf16* __restrict__ P,
                            int K, int C, int Kpad) {
  int ntilesC = C >> 4;
  long long total = (long long)(Kpad >> 5) * ntilesC * 512;
  long long i = (long long)blockIdx.x * blockDim.x + threadIdx.x;
  long long stride = (long long)gridDim.x * blockDim.x;
  for (; i < total; i += stride) {
    int blk  = (int)(i >> 9);
    int rem  = (int)(i & 511);
    int lane = rem >> 4;
    int e    = rem & 15;
    int kblk  = blk / ntilesC;
    int ntile = blk - kblk * ntilesC;
    int khalf = (lane >> 4) << 3;
    int kk  = (kblk << 5) + khalf + (e & 7) + ((e >> 3) << 4);
    int col = (ntile << 4) + (lane & 15);
    float v = (kk < K) ? W[(long long)kk * C + col] : 0.0f;
    P[i] = (__bf16)v;
  }
}

// ---------------------------------------------------------------- fragment loads (unconditional b128s)
// A: bf16 row-major [N, Kpad]; lane needs elements [khalf+kb, +8) and [khalf+kb+16, +8).
__device__ __forceinline__ v16bf rgcn_load_a(const __bf16* __restrict__ A,
                                             long long rowoff, int kb) {
  union { uint4 q[2]; v16bf bf; } f;
  const uint4* p = (const uint4*)(A + rowoff + kb);
  f.q[0] = p[0];     // 8 bf16: K = khalf+kb .. +7
  f.q[1] = p[2];     // 8 bf16: K = khalf+kb+16 .. +23
  return f.bf;
}

// B: pre-packed; lane reads its contiguous 32 bytes.
__device__ __forceinline__ v16bf rgcn_load_b(const __bf16* __restrict__ Wp,
                                             int kblk, int ntile, int ntilesC, int lane) {
  union { uint4 q[2]; v16bf bf; } f;
  const uint4* p = (const uint4*)(Wp + ((long long)(kblk * ntilesC + ntile) * 32 + lane) * 16);
  f.q[0] = p[0];
  f.q[1] = p[1];
  return f.bf;
}

// ---------------------------------------------------------------- dual GEMM + bias + BN stats
// Y = A1@W1 + A2@W2 + bias ; stats[c] += sum, stats[C+c] += sum^2
__global__ void rgcn_gemm_dual_stats(const __bf16* __restrict__ A1,
                                     const __bf16* __restrict__ A2,
                                     const __bf16* __restrict__ W1p,
                                     const __bf16* __restrict__ W2p,
                                     const float* __restrict__ bias,
                                     float* __restrict__ Y,
                                     float* __restrict__ stats,
                                     int Nrows, int Kpad, int C) {
  int wave = (int)((blockIdx.x * blockDim.x + threadIdx.x) >> 5);
  int ntilesC = C >> 4;
  int total_tiles = (Nrows >> 4) * ntilesC;
  if (wave >= total_tiles) return;           // wave-uniform: EXEC stays all-ones
  int mtile = wave / ntilesC;
  int ntile = wave - mtile * ntilesC;

  int lane  = threadIdx.x & 31;
  int lrow  = lane & 15;
  int khalf = (lane >> 4) << 3;              // 0 or 8
  int row   = (mtile << 4) + lrow;
  int col   = (ntile << 4) + lrow;
  long long rowoff = (long long)row * Kpad + khalf;

  v8f acc = {};
  int kblk = 0;
  for (int kb = 0; kb < Kpad; kb += 32, ++kblk) {
    v16bf a1 = rgcn_load_a(A1, rowoff, kb);
    v16bf b1 = rgcn_load_b(W1p, kblk, ntile, ntilesC, lane);
    acc = __builtin_amdgcn_wmma_f32_16x16x32_bf16(false, a1, false, b1,
                                                  (short)0, acc, false, false);
    v16bf a2 = rgcn_load_a(A2, rowoff, kb);
    v16bf b2 = rgcn_load_b(W2p, kblk, ntile, ntilesC, lane);
    acc = __builtin_amdgcn_wmma_f32_16x16x32_bf16(false, a2, false, b2,
                                                  (short)0, acc, false, false);
  }

  // C/D layout: VGPR r -> M = r (lanes 0-15) or 8+r (lanes 16-31); N = lane&15.
  float bv = bias[col];
  float s = 0.0f, s2 = 0.0f;
  int mbase = (mtile << 4) + khalf;
#pragma unroll
  for (int r = 0; r < 8; ++r) {
    float v = acc[r] + bv;
    Y[(long long)(mbase + r) * C + col] = v;
    s += v;
    s2 += v * v;
  }
  __hip_atomic_fetch_add(&stats[col],     s,  __ATOMIC_RELAXED, __HIP_MEMORY_SCOPE_AGENT);
  __hip_atomic_fetch_add(&stats[C + col], s2, __ATOMIC_RELAXED, __HIP_MEMORY_SCOPE_AGENT);
}

// ---------------------------------------------------------------- GEMM + bias + ReLU (head)
__global__ void rgcn_gemm_bias_relu(const __bf16* __restrict__ A,
                                    const __bf16* __restrict__ Wp,
                                    const float* __restrict__ bias,
                                    float* __restrict__ Y,
                                    int Nrows, int Kpad, int C) {
  int wave = (int)((blockIdx.x * blockDim.x + threadIdx.x) >> 5);
  int ntilesC = C >> 4;
  int total_tiles = (Nrows >> 4) * ntilesC;
  if (wave >= total_tiles) return;
  int mtile = wave / ntilesC;
  int ntile = wave - mtile * ntilesC;

  int lane  = threadIdx.x & 31;
  int lrow  = lane & 15;
  int khalf = (lane >> 4) << 3;
  int row   = (mtile << 4) + lrow;
  int col   = (ntile << 4) + lrow;
  long long rowoff = (long long)row * Kpad + khalf;

  v8f acc = {};
  int kblk = 0;
  for (int kb = 0; kb < Kpad; kb += 32, ++kblk) {
    v16bf a = rgcn_load_a(A, rowoff, kb);
    v16bf b = rgcn_load_b(Wp, kblk, ntile, ntilesC, lane);
    acc = __builtin_amdgcn_wmma_f32_16x16x32_bf16(false, a, false, b,
                                                  (short)0, acc, false, false);
  }

  float bv = bias[col];
  int mbase = (mtile << 4) + khalf;
#pragma unroll
  for (int r = 0; r < 8; ++r) {
    float v = acc[r] + bv;
    Y[(long long)(mbase + r) * C + col] = fmaxf(v, 0.0f);
  }
}

// ---------------------------------------------------------------- batch-norm + ReLU, dual output
// H (fp32, for next scatter) and Hbf (bf16 packed, for next GEMM A operand).
__global__ void rgcn_bn_relu_dual(const float* __restrict__ Y,
                                  const float* __restrict__ stats,
                                  const float* __restrict__ gamma,
                                  const float* __restrict__ beta,
                                  float* __restrict__ H,
                                  __bf16* __restrict__ Hbf,
                                  long long totalPairs, int C) {
  const float invN = 1.0f / (float)NN;
  long long idx = (long long)blockIdx.x * blockDim.x + threadIdx.x;
  long long stride = (long long)gridDim.x * blockDim.x;
  unsigned int* __restrict__ Hbp = (unsigned int*)Hbf;
  for (; idx < totalPairs; idx += stride) {
    long long i = idx << 1;
    int c = (int)(i % C);
    float m0 = stats[c]     * invN;
    float m1 = stats[c + 1] * invN;
    float va0 = stats[C + c]     * invN - m0 * m0;
    float va1 = stats[C + c + 1] * invN - m1 * m1;
    float v0 = (Y[i]     - m0) * rsqrtf(va0 + BN_EPS) * gamma[c]     + beta[c];
    float v1 = (Y[i + 1] - m1) * rsqrtf(va1 + BN_EPS) * gamma[c + 1] + beta[c + 1];
    v0 = fmaxf(v0, 0.0f);
    v1 = fmaxf(v1, 0.0f);
    H[i]     = v0;
    H[i + 1] = v1;
    union { unsigned int u; __bf16 h[2]; } pk;
    pk.h[0] = (__bf16)v0;
    pk.h[1] = (__bf16)v1;
    Hbp[idx] = pk.u;
  }
}

// ---------------------------------------------------------------- final 128 -> 1 projection
__global__ void rgcn_head2(const float* __restrict__ H,
                           const float* __restrict__ w,
                           const float* __restrict__ b,
                           float* __restrict__ out, int Nrows) {
  int wave = (int)((blockIdx.x * blockDim.x + threadIdx.x) >> 5);
  int lane = threadIdx.x & 31;
  if (wave >= Nrows) return;
  const float* hp = H + (long long)wave * 128;
  float s = 0.0f;
#pragma unroll
  for (int c = 0; c < 128; c += 32) s += hp[c + lane] * w[c + lane];
  for (int off = 16; off > 0; off >>= 1) s += __shfl_down(s, off, 32);
  if (lane == 0) out[wave] = s + b[0];
}

// ---------------------------------------------------------------- host-side launch helpers
static void launch_fill(float* p, long long n, hipStream_t stream) {
  int blocks = (int)llmin_host((n + 255) / 256, 65535);
  rgcn_fill0<<<blocks, 256, 0, stream>>>(p, n);
}

static void launch_scatter(const float* x, const long long* src, const long long* dst,
                           float* agg, int C, hipStream_t stream) {
  long long total = (long long)NE * C;
  int blocks = (int)llmin_host((total + 255) / 256, 131072);
  rgcn_scatter_add<<<blocks, 256, 0, stream>>>(x, src, dst, agg, total, C);
}

static void launch_cvt(const float* X, __bf16* Y, int K, int Kpad, hipStream_t stream) {
  long long pairs = (long long)NN * (Kpad >> 1);
  int blocks = (int)llmin_host((pairs + 255) / 256, 65535);
  rgcn_cvt_pad_bf16<<<blocks, 256, 0, stream>>>(X, Y, NN, K, Kpad);
}

static void launch_pack_w(const float* W, __bf16* P, int K, int C, int Kpad,
                          hipStream_t stream) {
  long long total = (long long)(Kpad >> 5) * (C >> 4) * 512;
  int blocks = (int)llmin_host((total + 255) / 256, 4096);
  rgcn_pack_w<<<blocks, 256, 0, stream>>>(W, P, K, C, Kpad);
}

static void launch_gemm_dual(const __bf16* A1, const __bf16* A2, const __bf16* W1p,
                             const __bf16* W2p, const float* bias, float* Y, float* stats,
                             int Kpad, int C, hipStream_t stream) {
  long long tiles = (long long)(NN / 16) * (C / 16);
  int blocks = (int)((tiles + 7) / 8);          // 8 waves (256 thr) per block
  rgcn_gemm_dual_stats<<<blocks, 256, 0, stream>>>(A1, A2, W1p, W2p, bias, Y, stats,
                                                   NN, Kpad, C);
}

// packed-weight blob size in bf16 elements
static inline long long wp_elems(int Kpad, int C) {
  return (long long)(Kpad / 32) * (C / 16) * 512;
}

extern "C" void kernel_launch(void* const* d_in, const int* in_sizes, int n_in,
                              void* d_out, int out_size, void* d_ws, size_t ws_size,
                              hipStream_t stream) {
  (void)in_sizes; (void)n_in; (void)out_size; (void)ws_size;

  const float*     x    = (const float*)d_in[0];
  const long long* ei   = (const long long*)d_in[1];
  const long long* src  = ei;            // edge_index[0]
  const long long* dstp = ei + NE;       // edge_index[1]

  const float* w_rel1  = (const float*)d_in[2];
  const float* w_root1 = (const float*)d_in[3];
  const float* b1      = (const float*)d_in[4];
  const float* gamma1  = (const float*)d_in[5];
  const float* beta1   = (const float*)d_in[6];
  const float* w_rel2  = (const float*)d_in[7];
  const float* w_root2 = (const float*)d_in[8];
  const float* b2      = (const float*)d_in[9];
  const float* gamma2  = (const float*)d_in[10];
  const float* beta2   = (const float*)d_in[11];
  const float* w_rel3  = (const float*)d_in[12];
  const float* w_root3 = (const float*)d_in[13];
  const float* b3      = (const float*)d_in[14];
  const float* gamma3  = (const float*)d_in[15];
  const float* beta3   = (const float*)d_in[16];
  const float* w_h1    = (const float*)d_in[17];
  const float* b_h1    = (const float*)d_in[18];
  const float* w_h2    = (const float*)d_in[19];
  const float* b_h2    = (const float*)d_in[20];

  // ---------------- workspace carve-up
  float* ws    = (float*)d_ws;
  float* agg   = ws;                                  // [NN,128] f32 max
  float* y     = agg  + (size_t)NN * 128;             // [NN,256] f32 (pre-BN)
  float* hA    = y    + (size_t)NN * 256;             // activations ping (f32)
  float* hB    = hA   + (size_t)NN * 256;             // activations pong (f32)
  float* stats = hB   + (size_t)NN * 256;             // [2*256]
  __bf16* bfb  = (__bf16*)(stats + 512);
  __bf16* xbf    = bfb;                               // [NN,32]  (x padded 6->32)
  __bf16* aggbf  = xbf   + (size_t)NN * 32;           // [NN,128] max
  __bf16* hAbf   = aggbf + (size_t)NN * 128;          // [NN,256]
  __bf16* hBbf   = hAbf  + (size_t)NN * 256;          // [NN,256]
  __bf16* wp     = hBbf  + (size_t)NN * 256;          // packed weights
  __bf16* wpRel1  = wp;
  __bf16* wpRoot1 = wpRel1  + wp_elems(32, 64);
  __bf16* wpRel2  = wpRoot1 + wp_elems(32, 64);
  __bf16* wpRoot2 = wpRel2  + wp_elems(64, 128);
  __bf16* wpRel3  = wpRoot2 + wp_elems(64, 128);
  __bf16* wpRoot3 = wpRel3  + wp_elems(128, 256);
  __bf16* wpH1    = wpRoot3 + wp_elems(128, 256);
  float* out   = (float*)d_out;

  long long bn_pairs;
  int bn_blocks;

  // ---------------- pack all weights (tiny) + convert x
  launch_pack_w(w_rel1,  wpRel1,  6,   64,  32,  stream);
  launch_pack_w(w_root1, wpRoot1, 6,   64,  32,  stream);
  launch_pack_w(w_rel2,  wpRel2,  64,  128, 64,  stream);
  launch_pack_w(w_root2, wpRoot2, 64,  128, 64,  stream);
  launch_pack_w(w_rel3,  wpRel3,  128, 256, 128, stream);
  launch_pack_w(w_root3, wpRoot3, 128, 256, 128, stream);
  launch_pack_w(w_h1,    wpH1,    256, 128, 256, stream);
  launch_cvt(x, xbf, 6, 32, stream);

  // ---- Layer 1: x [NN,6] -> h1 [NN,64] (hA / hAbf)
  launch_fill(agg, (long long)NN * 6, stream);
  launch_fill(stats, 2 * 64, stream);
  launch_scatter(x, src, dstp, agg, 6, stream);
  launch_cvt(agg, aggbf, 6, 32, stream);
  launch_gemm_dual(aggbf, xbf, wpRel1, wpRoot1, b1, y, stats, 32, 64, stream);
  bn_pairs = (long long)NN * 64 / 2;
  bn_blocks = (int)llmin_host((bn_pairs + 255) / 256, 65535);
  rgcn_bn_relu_dual<<<bn_blocks, 256, 0, stream>>>(y, stats, gamma1, beta1,
                                                   hA, hAbf, bn_pairs, 64);

  // ---- Layer 2: h1 [NN,64] -> h2 [NN,128] (hB / hBbf)
  launch_fill(agg, (long long)NN * 64, stream);
  launch_fill(stats, 2 * 128, stream);
  launch_scatter(hA, src, dstp, agg, 64, stream);
  launch_cvt(agg, aggbf, 64, 64, stream);
  launch_gemm_dual(aggbf, hAbf, wpRel2, wpRoot2, b2, y, stats, 64, 128, stream);
  bn_pairs = (long long)NN * 128 / 2;
  bn_blocks = (int)llmin_host((bn_pairs + 255) / 256, 65535);
  rgcn_bn_relu_dual<<<bn_blocks, 256, 0, stream>>>(y, stats, gamma2, beta2,
                                                   hB, hBbf, bn_pairs, 128);

  // ---- Layer 3: h2 [NN,128] -> h3 [NN,256] (hA / hAbf)
  launch_fill(agg, (long long)NN * 128, stream);
  launch_fill(stats, 2 * 256, stream);
  launch_scatter(hB, src, dstp, agg, 128, stream);
  launch_cvt(agg, aggbf, 128, 128, stream);
  launch_gemm_dual(aggbf, hBbf, wpRel3, wpRoot3, b3, y, stats, 128, 256, stream);
  bn_pairs = (long long)NN * 256 / 2;
  bn_blocks = (int)llmin_host((bn_pairs + 255) / 256, 65535);
  rgcn_bn_relu_dual<<<bn_blocks, 256, 0, stream>>>(y, stats, gamma3, beta3,
                                                   hA, hAbf, bn_pairs, 256);

  // ---- Head: h4 = relu(h3 @ w_h1 + b_h1)   [NN,128] fp32 into hB
  {
    long long tiles = (long long)(NN / 16) * (128 / 16);
    int blocks = (int)((tiles + 7) / 8);
    rgcn_gemm_bias_relu<<<blocks, 256, 0, stream>>>(hAbf, wpH1, b_h1, hB,
                                                    NN, 256, 128);
  }

  // ---- out = h4 @ w_h2 + b_h2   (one wave per node)
  {
    int blocks = (NN * 32 + 255) / 256;
    rgcn_head2<<<blocks, 256, 0, stream>>>(hB, w_h2, b_h2, out, NN);
  }
}
